// GNNEncoder_83056077570951
// MI455X (gfx1250) — matrix operand order repaired and checked by reading, not compiled, not run on hardware
//
#include <hip/hip_runtime.h>
#include <math.h>

typedef _Float16 half_t;
typedef __attribute__((ext_vector_type(2)))  _Float16 h2;
typedef __attribute__((ext_vector_type(16))) _Float16 v16h;
typedef __attribute__((ext_vector_type(8)))  float    v8f;

#define BM 128
#define BK 32

// epilogues
#define EPI_NONE      0
#define EPI_RELU      1
#define EPI_SKIP      3   // gated skip blend (conv2 out)
#define EPI_SKIPLEAKY 4   // gated skip blend + leaky relu (conv1 out)

// ---------------------------------------------------------------------------
// Tiled f16 WMMA GEMM:  Y[N,M] = epi( A[N,K] @ B[K,M] + bias )
// Template CT = number of 16-wide col subtiles per wave (2 -> BN=64, 4 -> BN=128).
// 8 waves arranged 4 row-groups x 2 col-groups; wave tile = 32 x (CT*16).
// grid.x = row tiles (128), grid.y = col tiles (BN), grid.z = heads.
// Per-head offsets hsA/hsB/hsO (elements) support block-diagonal per-head
// 64x64 relation GEMMs with the same kernel.
// ---------------------------------------------------------------------------
template<int CT>
__global__ __launch_bounds__(256) void gemm_wmma(
    const half_t* __restrict__ A, const half_t* __restrict__ B,
    const float* __restrict__ bias,
    half_t* __restrict__ out16, float* __restrict__ out32,
    const half_t* __restrict__ skip, const float* __restrict__ gate,
    int N, int K, int lda, int ldb,
    int hsA, int hsB, int hsO, int epi)
{
    constexpr int BNt = 32 * CT;

    const int z = blockIdx.z;
    A += (long)z * hsA;
    B += (long)z * hsB;
    const long oOff = (long)z * hsO;

    const int rowBase = blockIdx.x * BM;
    const int colBase = blockIdx.y * BNt;

    __shared__ half_t As[BM * BK];    // [row][k]
    __shared__ half_t Bs[BNt * BK];   // [col][k]  (transposed at fill)

    const int t    = threadIdx.x;
    const int lane = t & 31;
    const int wid  = t >> 5;
    const int wr   = wid & 3;        // 4 row groups of 32 rows
    const int wc   = wid >> 2;       // 2 col groups of CT*16 cols
    const int g    = lane >> 4;      // lane half-group
    const int nidx = lane & 15;

    v8f acc[2][CT] = {};

    const int arow  = t >> 1;              // 0..127
    const int aoff  = (t & 1) * 16;        // 0 or 16
    const int brow  = t >> 3;              // k 0..31
    const int bcol0 = (t & 7) * (BNt / 8); // per-thread col segment

    for (int k0 = 0; k0 < K; k0 += BK) {
        // ---- stage A tile (row major), zero-fill OOB rows ----
        {
            const int grow = rowBase + arow;
            uint4 va = {0u, 0u, 0u, 0u}, vb = {0u, 0u, 0u, 0u};
            if (grow < N) {
                const uint4* src = (const uint4*)(A + (long)grow * lda + k0 + aoff);
                va = src[0];
                vb = src[1];
            }
            uint4* dst = (uint4*)&As[arow * BK + aoff];
            dst[0] = va;
            dst[1] = vb;
        }
        // ---- stage B tile transposed: Bs[col][k] ----
        {
            const half_t* bsrc = B + (long)(k0 + brow) * ldb + colBase + bcol0;
#pragma unroll
            for (int u = 0; u < CT / 2; u++) {
                const uint4 v = ((const uint4*)bsrc)[u];
                const half_t* hv = (const half_t*)&v;
#pragma unroll
                for (int j = 0; j < 8; j++)
                    Bs[(bcol0 + u * 8 + j) * BK + brow] = hv[j];
            }
        }
        __syncthreads();

        // ---- build fragments per ISA 7.12.2 layouts ----
        union Frag { v16h v; h2 p[8]; };
        Frag af[2], bf[CT];
#pragma unroll
        for (int rr = 0; rr < 2; rr++) {
            const int r = wr * 32 + rr * 16 + nidx;
#pragma unroll
            for (int p = 0; p < 8; p++) {
                const int kb = 2 * p + 8 * g + ((p >= 4) ? 8 : 0); // A: K pair index
                af[rr].p[p] = *(const h2*)&As[r * BK + kb];
            }
        }
#pragma unroll
        for (int cc = 0; cc < CT; cc++) {
            const int c = wc * (CT * 16) + cc * 16 + nidx;
            const h2* src = (const h2*)&Bs[c * BK + 8 * g];   // B: K pair index same as A
#pragma unroll
            for (int p = 0; p < 8; p++) bf[cc].p[p] = src[p + ((p >= 4) ? 4 : 0)];
        }
#pragma unroll
        for (int rr = 0; rr < 2; rr++)
#pragma unroll
            for (int cc = 0; cc < CT; cc++)
                acc[rr][cc] = __builtin_amdgcn_wmma_f32_16x16x32_f16(
                    false, af[rr].v, false, bf[cc].v,
                    (short)0, acc[rr][cc], false, false);
        __syncthreads();
    }

    float s = 0.f;
    if (epi == EPI_SKIP || epi == EPI_SKIPLEAKY)
        s = 1.f / (1.f + __expf(-gate[0]));

#pragma unroll
    for (int rr = 0; rr < 2; rr++) {
#pragma unroll
        for (int cc = 0; cc < CT; cc++) {
            const int c = colBase + wc * (CT * 16) + cc * 16 + nidx;
            const float bv = bias ? bias[c] : 0.f;
#pragma unroll
            for (int r8 = 0; r8 < 8; r8++) {
                const int row = rowBase + wr * 32 + rr * 16 + r8 + 8 * g; // C: M = r + 8g
                if (row >= N) continue;
                float v = acc[rr][cc][r8] + bv;
                if (epi == EPI_RELU) {
                    v = v > 0.f ? v : 0.f;
                } else if (epi == EPI_SKIP || epi == EPI_SKIPLEAKY) {
                    const float sk = (float)skip[(long)row * 256 + c];
                    v = s * v + (1.f - s) * sk;
                    if (epi == EPI_SKIPLEAKY) v = v > 0.f ? v : 0.01f * v;
                }
                const long oi = (long)row * 256 + c + oOff;
                if (out16) out16[oi] = (half_t)v;
                if (out32) out32[oi] = v;
            }
        }
    }
}

// ---------------------------------------------------------------------------
// small utility kernels
// ---------------------------------------------------------------------------
__global__ void cvt_f32_f16(const float* __restrict__ s, half_t* __restrict__ d, int n)
{
    const int i = blockIdx.x * 256 + threadIdx.x;
    if (i < n) d[i] = (half_t)s[i];
}

// arel [4][64][64] (h,d,f) -> f16 transposed per head (f,d), so qt = q @ arelT
__global__ void cvt_arelT(const float* __restrict__ s, half_t* __restrict__ d)
{
    const int i = blockIdx.x * 256 + threadIdx.x;
    if (i < 16384) {
        const int h = i >> 12, r = (i >> 6) & 63, c = i & 63;
        d[(h << 12) + (c << 6) + r] = (half_t)s[i];
    }
}

__global__ void fill_f32k(float* __restrict__ p, float v, long n)
{
    const long i = (long)blockIdx.x * 256 + threadIdx.x;
    if (i < n) p[i] = v;
}

__global__ void gelu_f32_f16(const float* __restrict__ s, half_t* __restrict__ d, long n)
{
    const long i = (long)blockIdx.x * 256 + threadIdx.x;
    if (i < n) {
        const float x = s[i];
        d[i] = (half_t)(0.5f * x * (1.f + erff(x * 0.70710678118f)));
    }
}

__device__ inline void atomicMaxF(float* addr, float val)
{
    unsigned int* a = (unsigned int*)addr;
    unsigned int old = *a;
    while (true) {
        const float f = __uint_as_float(old);
        if (f >= val) break;
        const unsigned int assumed = old;
        old = atomicCAS(a, assumed, __float_as_uint(val));
        if (old == assumed) break;
    }
}

// a[e,h] = dot64(k[si[e]][h], qt[di[e]][h]) * prel[h] * (1/sqrt(64))
// one wave32 per edge: lane covers 8 contiguous halves; 8 lanes per head.
__global__ __launch_bounds__(256) void edge_logits(
    const half_t* __restrict__ k, const half_t* __restrict__ qt,
    const int* __restrict__ si, const int* __restrict__ di,
    const float* __restrict__ prel, float* __restrict__ out, int E)
{
    const int lane = threadIdx.x & 31;
    const long e = (long)blockIdx.x * 8 + (threadIdx.x >> 5);
    if (e >= E) return;
    const long s = si[e], d = di[e];
    const h2* kr = (const h2*)(k + s * 256) + lane * 4;
    const h2* qr = (const h2*)(qt + d * 256) + lane * 4;
    float acc = 0.f;
#pragma unroll
    for (int j = 0; j < 4; j++) {
        const h2 a = kr[j], b = qr[j];
        acc += (float)a[0] * (float)b[0] + (float)a[1] * (float)b[1];
    }
    acc += __shfl_xor(acc, 1, 32);
    acc += __shfl_xor(acc, 2, 32);
    acc += __shfl_xor(acc, 4, 32);
    const int h = lane >> 3;
    if ((lane & 7) == 0) out[e * 4 + h] = acc * prel[h] * 0.125f;
}

__global__ void edge_amax(const float* __restrict__ a, const int* __restrict__ di,
                          float* __restrict__ amax, long n)
{
    const long i = (long)blockIdx.x * 256 + threadIdx.x;
    if (i >= n) return;
    const long e = i >> 2;
    const int h = (int)(i & 3);
    atomicMaxF(&amax[(long)di[e] * 4 + h], a[i]);
}

__global__ void edge_expsum(float* __restrict__ a, const int* __restrict__ di,
                            const float* __restrict__ amax, float* __restrict__ den, long n)
{
    const long i = (long)blockIdx.x * 256 + threadIdx.x;
    if (i >= n) return;
    const long e = i >> 2;
    const int h = (int)(i & 3);
    const float ex = __expf(a[i] - amax[(long)di[e] * 4 + h]);
    a[i] = ex;
    atomicAdd(&den[(long)di[e] * 4 + h], ex);
}

// agg[di] += alpha[e,h] * vt[si];  wave32 per edge, lane handles 8 channels
__global__ __launch_bounds__(256) void edge_scatter(
    const float* __restrict__ ex, const float* __restrict__ den,
    const half_t* __restrict__ vt, const int* __restrict__ si,
    const int* __restrict__ di, float* __restrict__ agg, int E)
{
    const int lane = threadIdx.x & 31;
    const long e = (long)blockIdx.x * 8 + (threadIdx.x >> 5);
    if (e >= E) return;
    const long s = si[e], d = di[e];
    const int h = lane >> 3;
    const float alpha = ex[e * 4 + h] / (den[d * 4 + h] + 1e-16f);
    const h2* vr = (const h2*)(vt + s * 256) + lane * 4;
    float* ag = agg + d * 256 + lane * 8;
#pragma unroll
    for (int j = 0; j < 4; j++) {
        const h2 v = vr[j];
        atomicAdd(&ag[2 * j],     (float)v[0] * alpha);
        atomicAdd(&ag[2 * j + 1], (float)v[1] * alpha);
    }
}

// ---------------------------------------------------------------------------
extern "C" void kernel_launch(void* const* d_in, const int* in_sizes, int n_in,
                              void* d_out, int out_size, void* d_ws, size_t ws_size,
                              hipStream_t stream)
{
    const int NC = 50000, NP = 20000, E = 300000;
    (void)in_sizes; (void)n_in; (void)out_size; (void)ws_size;

    auto F = [&](int i) { return (const float*)d_in[i]; };
    const int* eb_src = (const int*)d_in[2];
    const int* eb_dst = (const int*)d_in[3];
    const int* er_src = (const int*)d_in[4];
    const int* er_dst = (const int*)d_in[5];

    // param index helpers (flattened make_params order, after 6 data inputs)
    // 6:Win_c 7:bin_c 8:Win_p 9:bin_p, then per conv (24 entries):
    //   per nt: Wk Wq Wv Wo bk bq bv bo skip ; then per edge: arel mrel prel
    auto pconv = [&](int cv) { return 10 + cv * 24; };

    // ---- workspace layout ----
    char* wsb = (char*)d_ws;
    size_t off = 0;
    auto AH = [&](size_t n) -> half_t* {
        half_t* p = (half_t*)(wsb + off);
        off += ((n * sizeof(half_t) + 255) & ~(size_t)255);
        return p;
    };
    auto AF = [&](size_t n) -> float* {
        float* p = (float*)(wsb + off);
        off += ((n * sizeof(float) + 255) & ~(size_t)255);
        return p;
    };

    half_t* xc16 = AH((size_t)NC * 64);
    half_t* xp16 = AH((size_t)NP * 64);
    half_t* wWin_c = AH(64 * 256);
    half_t* wWin_p = AH(64 * 256);
    half_t* wW[2][2][4];       // [conv][nt][Wk,Wq,Wv,Wo]
    half_t* arelT16[2][2];     // [conv][buys,rev]
    half_t* mrel16[2][2];
    for (int cv = 0; cv < 2; cv++) {
        for (int nt = 0; nt < 2; nt++)
            for (int w = 0; w < 4; w++) wW[cv][nt][w] = AH(256 * 256);
        for (int et = 0; et < 2; et++) {
            arelT16[cv][et] = AH(4 * 64 * 64);
            mrel16[cv][et]  = AH(4 * 64 * 64);
        }
    }
    half_t* wWp1 = AH(256 * 256);
    half_t* wWp2 = AH(256 * 256);

    half_t* hc[2] = { AH((size_t)NC * 256), AH((size_t)NC * 256) };
    half_t* hp[2] = { AH((size_t)NP * 256), AH((size_t)NP * 256) };
    half_t* kc = AH((size_t)NC * 256);
    half_t* qc = AH((size_t)NC * 256);     // reused as gelu(agg_c)
    half_t* vc = AH((size_t)NC * 256);
    half_t* kp = AH((size_t)NP * 256);     // kc reused as proj intermediate
    half_t* qp = AH((size_t)NP * 256);
    half_t* vp = AH((size_t)NP * 256);
    half_t* qtc = AH((size_t)NC * 256);
    half_t* vtc = AH((size_t)NC * 256);
    half_t* qtp = AH((size_t)NP * 256);
    half_t* vtp = AH((size_t)NP * 256);

    float* aggc  = AF((size_t)NC * 256);
    float* aggp  = AF((size_t)NP * 256);
    float* amaxc = AF((size_t)NC * 4);
    float* denc  = AF((size_t)NC * 4);
    float* amaxp = AF((size_t)NP * 4);
    float* denp  = AF((size_t)NP * 4);
    float* exP   = AF((size_t)E * 4);   // logits->probs for edges into p (buys)
    float* exC   = AF((size_t)E * 4);   // edges into c (rev)

    // ---- launch helpers ----
    auto cvt = [&](const float* s, half_t* d, int n) {
        cvt_f32_f16<<<(n + 255) / 256, 256, 0, stream>>>(s, d, n);
    };
    auto fill = [&](float* p, float v, long n) {
        fill_f32k<<<(int)((n + 255) / 256), 256, 0, stream>>>(p, v, n);
    };
    auto gemm = [&](const half_t* A, int lda, const half_t* B, int ldb,
                    const float* bias, half_t* o16, float* o32,
                    const half_t* skip, const float* gate,
                    int N, int K, int M, int heads,
                    int hsA, int hsB, int hsO, int epi) {
        if (M % 128 == 0) {
            dim3 g((N + BM - 1) / BM, M / 128, heads);
            gemm_wmma<4><<<g, 256, 0, stream>>>(A, B, bias, o16, o32, skip, gate,
                                                N, K, lda, ldb, hsA, hsB, hsO, epi);
        } else {
            dim3 g((N + BM - 1) / BM, M / 64, heads);
            gemm_wmma<2><<<g, 256, 0, stream>>>(A, B, bias, o16, o32, skip, gate,
                                                N, K, lda, ldb, hsA, hsB, hsO, epi);
        }
    };

    // ---- weight/input conversion ----
    cvt(F(0), xc16, NC * 64);
    cvt(F(1), xp16, NP * 64);
    cvt(F(6), wWin_c, 64 * 256);
    cvt(F(8), wWin_p, 64 * 256);
    for (int cv = 0; cv < 2; cv++) {
        const int pb = pconv(cv);
        for (int nt = 0; nt < 2; nt++)
            for (int w = 0; w < 4; w++)
                cvt(F(pb + nt * 9 + w), wW[cv][nt][w], 256 * 256);
        for (int et = 0; et < 2; et++) {
            cvt_arelT<<<64, 256, 0, stream>>>(F(pb + 18 + et * 3), arelT16[cv][et]);
            cvt(F(pb + 18 + et * 3 + 1), mrel16[cv][et], 4 * 64 * 64);
        }
    }
    cvt(F(58), wWp1, 256 * 256);
    cvt(F(60), wWp2, 256 * 256);

    // ---- input projection: h = relu(x @ Win + bin) ----
    gemm(xc16, 64, wWin_c, 256, F(7), hc[0], nullptr, nullptr, nullptr,
         NC, 64, 256, 1, 0, 0, 0, EPI_RELU);
    gemm(xp16, 64, wWin_p, 256, F(9), hp[0], nullptr, nullptr, nullptr,
         NP, 64, 256, 1, 0, 0, 0, EPI_RELU);

    // ---- two HGT convs ----
    for (int cv = 0; cv < 2; cv++) {
        const int pb = pconv(cv);
        const half_t* inC = hc[cv & 1];
        const half_t* inP = hp[cv & 1];
        half_t* outC = hc[(cv + 1) & 1];
        half_t* outP = hp[(cv + 1) & 1];
        const int epiOut = (cv == 0) ? EPI_SKIPLEAKY : EPI_SKIP;
        float* out32c = (cv == 0) ? nullptr : (float*)d_out;  // emb

        // K/Q/V per node type
        for (int nt = 0; nt < 2; nt++) {
            const half_t* X = nt ? inP : inC;
            const int Nn = nt ? NP : NC;
            half_t* dst[3] = { nt ? kp : kc, nt ? qp : qc, nt ? vp : vc };
            for (int w = 0; w < 3; w++)
                gemm(X, 256, wW[cv][nt][w], 256, F(pb + nt * 9 + 4 + w),
                     dst[w], nullptr, nullptr, nullptr,
                     Nn, 256, 256, 1, 0, 0, 0, EPI_NONE);
        }
        // per-head relation transforms (4 heads, block-diagonal 64x64 GEMMs)
        // qt_c: dst of 'rev'  -> arel_rev^T ; qt_p: dst of 'buys' -> arel_buys^T
        // vt_c: src of 'buys' -> mrel_buys  ; vt_p: src of 'rev'  -> mrel_rev
        gemm(qc, 256, arelT16[cv][1], 64, nullptr, qtc, nullptr, nullptr, nullptr,
             NC, 64, 64, 4, 64, 4096, 64, EPI_NONE);
        gemm(qp, 256, arelT16[cv][0], 64, nullptr, qtp, nullptr, nullptr, nullptr,
             NP, 64, 64, 4, 64, 4096, 64, EPI_NONE);
        gemm(vc, 256, mrel16[cv][0], 64, nullptr, vtc, nullptr, nullptr, nullptr,
             NC, 64, 64, 4, 64, 4096, 64, EPI_NONE);
        gemm(vp, 256, mrel16[cv][1], 64, nullptr, vtp, nullptr, nullptr, nullptr,
             NP, 64, 64, 4, 64, 4096, 64, EPI_NONE);

        // softmax state init
        fill(amaxc, -1e30f, (long)NC * 4);
        fill(amaxp, -1e30f, (long)NP * 4);
        fill(denc, 0.f, (long)NC * 4);
        fill(denp, 0.f, (long)NP * 4);
        fill(aggc, 0.f, (long)NC * 256);
        fill(aggp, 0.f, (long)NP * 256);

        const int egrid = (E + 7) / 8;
        const int agrid = (int)(((long)E * 4 + 255) / 256);
        // logits
        edge_logits<<<egrid, 256, 0, stream>>>(kc, qtp, eb_src, eb_dst,
                                               F(pb + 18 + 2), exP, E);         // buys -> p
        edge_logits<<<egrid, 256, 0, stream>>>(kp, qtc, er_src, er_dst,
                                               F(pb + 18 + 3 + 2), exC, E);     // rev  -> c
        // segment max
        edge_amax<<<agrid, 256, 0, stream>>>(exP, eb_dst, amaxp, (long)E * 4);
        edge_amax<<<agrid, 256, 0, stream>>>(exC, er_dst, amaxc, (long)E * 4);
        // exp + segment sum
        edge_expsum<<<agrid, 256, 0, stream>>>(exP, eb_dst, amaxp, denp, (long)E * 4);
        edge_expsum<<<agrid, 256, 0, stream>>>(exC, er_dst, amaxc, denc, (long)E * 4);
        // weighted aggregation
        edge_scatter<<<egrid, 256, 0, stream>>>(exP, denp, vtc, eb_src, eb_dst, aggp, E);
        edge_scatter<<<egrid, 256, 0, stream>>>(exC, denc, vtp, er_src, er_dst, aggc, E);

        // gelu(agg) -> f16  (reuse q buffers)
        gelu_f32_f16<<<(int)(((long)NC * 256 + 255) / 256), 256, 0, stream>>>(aggc, qc, (long)NC * 256);
        gelu_f32_f16<<<(int)(((long)NP * 256 + 255) / 256), 256, 0, stream>>>(aggp, qp, (long)NP * 256);

        // out = s * (gelu @ Wo + bo) + (1-s) * h_in   (+ leaky for conv1)
        gemm(qc, 256, wW[cv][0][3], 256, F(pb + 0 * 9 + 7), outC, out32c,
             inC, F(pb + 0 * 9 + 8), NC, 256, 256, 1, 0, 0, 0, epiOut);
        gemm(qp, 256, wW[cv][1][3], 256, F(pb + 1 * 9 + 7), outP, nullptr,
             inP, F(pb + 1 * 9 + 8), NP, 256, 256, 1, 0, 0, 0, epiOut);
    }

    // ---- projection head on customers ----
    // t = relu(emb @ Wp1 + bp1) ; proj = t @ Wp2 + bp2
    half_t* emb16 = hc[0];   // conv2 wrote into hc[0]
    half_t* tproj = kc;      // reuse
    gemm(emb16, 256, wWp1, 256, F(59), tproj, nullptr, nullptr, nullptr,
         NC, 256, 256, 1, 0, 0, 0, EPI_RELU);
    gemm(tproj, 256, wWp2, 256, F(61), nullptr,
         ((float*)d_out) + (size_t)NC * 256, nullptr, nullptr,
         NC, 256, 256, 1, 0, 0, 0, EPI_NONE);
}
